// Conv3d_31739808317553
// MI455X (gfx1250) — matrix operand order repaired and checked by reading, not compiled, not run on hardware
//
#include <hip/hip_runtime.h>
#include <hip/hip_bf16.h>

typedef __attribute__((ext_vector_type(16))) __bf16 v16bf;
typedef __attribute__((ext_vector_type(8)))  float  v8f;
typedef int v4i __attribute__((vector_size(16)));

#define AS1 __attribute__((address_space(1)))
#define AS3 __attribute__((address_space(3)))

#if defined(__has_builtin)
#if __has_builtin(__builtin_amdgcn_global_load_async_to_lds_b128) && \
    __has_builtin(__builtin_amdgcn_s_wait_asynccnt)
#define USE_ASYNC 1
#else
#define USE_ASYNC 0
#endif
#else
#define USE_ASYNC 0
#endif

#define NVOX   262144
#define C      64
#define KVOL   27
#define PPO    65536
#define TILE   128      // pairs per block-iteration (8 waves x 16 rows)
#define WAVES  8
#define PBLK   64       // gridDim.x
#define TPB    8        // tiles per block: PBLK*TPB*TILE == PPO
#define NBUF   (USE_ASYNC ? 2 : 1)

__global__ void zero_f4(float4* out, int n4) {
    int i = blockIdx.x * blockDim.x + threadIdx.x;
    if (i < n4) out[i] = make_float4(0.f, 0.f, 0.f, 0.f);
}

__global__ __launch_bounds__(256) void
sparse_conv_wmma(const float* __restrict__ feat,
                 const float* __restrict__ kernel,
                 const int*   __restrict__ src_ids,
                 const int*   __restrict__ tgt_ids,
                 float*       __restrict__ out)
{
    const int k = blockIdx.y;
    const float* W   = kernel  + (long)k * C * C;   // [K=64][N=64]
    const int*   src = src_ids + (long)k * PPO;
    const int*   tgt = tgt_ids + (long)k * PPO;

    // B fragments pre-laid-out: [kc][nt][hi/lo][lane][elem]  (16 KB)
    __shared__ __align__(32) __bf16 wfrag[2][4][2][32][16];
    // gathered rows, fp32, padded against bank conflicts; double-buffered in async mode
    __shared__ float afeat[NBUF][WAVES][16][68];
    __shared__ int   atgt[NBUF][WAVES][16];

    const int tid  = threadIdx.x;
    const int wave = tid >> 5;
    const int lane = tid & 31;
    const int h    = lane >> 4;     // lane half
    const int lr   = lane & 15;

    // ---- split weights into hi/lo bf16, stored in B-fragment layout ----
    // B elem i at lane L: K = kc*32 + (L<16?0:16) + i, N = nt*16 + (L&15)
    for (int e = tid; e < 2 * 4 * 32 * 16; e += 256) {
        int i  = e & 15;
        int ln = (e >> 4) & 31;
        int nt = (e >> 9) & 3;
        int kc = (e >> 11) & 1;
        int Kg = kc * 32 + ((ln < 16) ? 0 : 16) + i;
        int Ng = nt * 16 + (ln & 15);
        float w   = W[Kg * C + Ng];
        __bf16 hi = (__bf16)w;
        __bf16 lo = (__bf16)(w - (float)hi);
        wfrag[kc][nt][0][ln][i] = hi;
        wfrag[kc][nt][1][ln][i] = lo;
    }
    __syncthreads();

    // Per-wave gather staging: each lane owns half a row (32 floats = 8 x b128)
    auto issue_stage = [&](int t, int buf) {
        int p0   = (blockIdx.x * TPB + t) * TILE + wave * 16;
        int row  = lane >> 1;
        int half = lane & 1;
        int sidx = src[p0 + row];
        const float* g = feat + (long)sidx * C + half * 32;
        float*       l = &afeat[buf][wave][row][half * 32];
#if USE_ASYNC
        #pragma unroll
        for (int j = 0; j < 8; ++j) {
            __builtin_amdgcn_global_load_async_to_lds_b128(
                (AS1 v4i*)(g + j * 4), (AS3 v4i*)(l + j * 4), 0, 0);
        }
#else
        #pragma unroll
        for (int j = 0; j < 8; ++j)
            *(float4*)(l + j * 4) = ((const float4*)g)[j];
#endif
        if (lane < 16) atgt[buf][wave][lane] = tgt[p0 + lane];
    };

    auto compute_tile = [&](int buf) {
        v8f acc[4] = {v8f{}, v8f{}, v8f{}, v8f{}};
        #pragma unroll
        for (int kc = 0; kc < 2; ++kc) {
            // A elem i at lane L: row = L&15, K = kc*32 + h*16 + i
            v16bf ahi, alo;
            #pragma unroll
            for (int i = 0; i < 16; ++i) {
                int Kg  = kc * 32 + (h << 4) + i;
                float x = afeat[buf][wave][lr][Kg];
                __bf16 hi = (__bf16)x;
                ahi[i] = hi;
                alo[i] = (__bf16)(x - (float)hi);
            }
            #pragma unroll
            for (int nt = 0; nt < 4; ++nt) {
                v16bf bhi = *(const v16bf*)&wfrag[kc][nt][0][lane][0];
                v16bf blo = *(const v16bf*)&wfrag[kc][nt][1][lane][0];
                acc[nt] = __builtin_amdgcn_wmma_f32_16x16x32_bf16(
                    false, ahi, false, bhi, (short)0, acc[nt], false, false);
                acc[nt] = __builtin_amdgcn_wmma_f32_16x16x32_bf16(
                    false, ahi, false, blo, (short)0, acc[nt], false, false);
                acc[nt] = __builtin_amdgcn_wmma_f32_16x16x32_bf16(
                    false, alo, false, bhi, (short)0, acc[nt], false, false);
            }
        }
        // scatter-add; C layout: lanes 0-15 -> M=j, lanes 16-31 -> M=j+8
        int trow[8];
        #pragma unroll
        for (int j = 0; j < 8; ++j) trow[j] = atgt[buf][wave][j + h * 8];
        #pragma unroll
        for (int nt = 0; nt < 4; ++nt) {
            #pragma unroll
            for (int j = 0; j < 8; ++j) {
                float* p = out + (long)trow[j] * C + nt * 16 + lr;
                __hip_atomic_fetch_add(p, acc[nt][j], __ATOMIC_RELAXED,
                                       __HIP_MEMORY_SCOPE_AGENT);
            }
        }
    };

#if USE_ASYNC
    // Software pipeline: gathers for tile t+1 fly (ASYNCcnt) under tile t's WMMAs.
    // All afeat/atgt traffic is intra-wave -> no block barriers needed.
    issue_stage(0, 0);
    for (int t = 0; t < TPB; ++t) {
        int nb = t & 1;
        if (t + 1 < TPB) issue_stage(t + 1, nb ^ 1);
        asm volatile("" ::: "memory");
        if (t + 1 < TPB) __builtin_amdgcn_s_wait_asynccnt(8);  // tile t done, t+1 in flight
        else             __builtin_amdgcn_s_wait_asynccnt(0);
        asm volatile("" ::: "memory");
        compute_tile(nb);
    }
#else
    for (int t = 0; t < TPB; ++t) {
        issue_stage(t, 0);
        __syncthreads();
        compute_tile(0);
        __syncthreads();
    }
#endif
}

extern "C" void kernel_launch(void* const* d_in, const int* in_sizes, int n_in,
                              void* d_out, int out_size, void* d_ws, size_t ws_size,
                              hipStream_t stream) {
    (void)in_sizes; (void)n_in; (void)d_ws; (void)ws_size;
    const float* feat = (const float*)d_in[0];
    const float* kern = (const float*)d_in[1];
    const int*   src  = (const int*)d_in[2];
    const int*   tgt  = (const int*)d_in[3];
    float*       out  = (float*)d_out;

    int n4 = out_size / 4;
    zero_f4<<<(n4 + 255) / 256, 256, 0, stream>>>((float4*)out, n4);

    dim3 grid(PBLK, KVOL);
    sparse_conv_wmma<<<grid, 256, 0, stream>>>(feat, kern, src, tgt, out);
}
